// message_passing_ablation_no_cau_24489903522214
// MI455X (gfx1250) — matrix operand order repaired and checked by reading, not compiled, not run on hardware
//
#include <hip/hip_runtime.h>
#include <hip/hip_bf16.h>
#include <math.h>

#define N_NODES 50000
#define N_EDGES 800000
#define DIM     128
#define N_HEADS 8
#define D_K     16

static constexpr size_t ND  = (size_t)N_NODES * DIM;   // 6,400,000
static constexpr size_t E8  = (size_t)N_EDGES * N_HEADS;
static constexpr size_t N8  = (size_t)N_NODES * N_HEADS;

typedef __attribute__((ext_vector_type(16))) _Float16 v16h;
typedef __attribute__((ext_vector_type(8)))  _Float16 v8h;
typedef __attribute__((ext_vector_type(8)))  float    v8f;

// ---- monotonic float<->uint encoding for atomicMax on signed floats ----
__device__ __forceinline__ unsigned enc_f32(float f) {
    unsigned u = __float_as_uint(f);
    return (u & 0x80000000u) ? ~u : (u | 0x80000000u);
}
__device__ __forceinline__ float dec_f32(unsigned u) {
    return (u & 0x80000000u) ? __uint_as_float(u & 0x7fffffffu)
                             : __uint_as_float(~u);
}

// ---------------------------------------------------------------- helpers
__global__ void k_f32_to_f16(const float* __restrict__ in,
                             _Float16* __restrict__ out, int n) {
    int i = blockIdx.x * blockDim.x + threadIdx.x;
    if (i < n) out[i] = (_Float16)in[i];
}

__global__ void k_zero_u32(unsigned* __restrict__ p, int n) {
    int i = blockIdx.x * blockDim.x + threadIdx.x;
    if (i < n) p[i] = 0u;
}

// ------------------------------------------------- fused k/q/v projections
// One wave per 16x16 output tile. D = A(16x32 f16) x B(32x16 f16) + C, K=128
// A[m,k] = xh[n0+m, k]   (row-major activations)
// B[k,n] = Wh[o0+n, k]   (PyTorch Linear: out = x @ W^T  ->  B = W^T)
__global__ void k_gemm_qkv(const _Float16* __restrict__ xh,
                           const _Float16* __restrict__ wh,   // [3][128][128]
                           const float* __restrict__ bk,
                           const float* __restrict__ bq,
                           const float* __restrict__ bv,
                           float* __restrict__ kqv)           // [3][N][128]
{
    int wid  = blockIdx.x * (blockDim.x >> 5) + (threadIdx.x >> 5);
    int lane = threadIdx.x & 31;
    int mat  = wid / 25000;            // 3125*8 tiles per matrix
    int t    = wid - mat * 25000;
    int tm   = t >> 3;                 // node tile 0..3124
    int tn   = t & 7;                  // out-col tile 0..7
    int n0   = tm * 16, o0 = tn * 16;
    int ml   = lane & 15;
    int hi   = lane >> 4;              // lane group (ISA A/B layout split)

    const _Float16* wm   = wh + (size_t)mat * DIM * DIM;
    const float*    bias = (mat == 0) ? bk : (mat == 1) ? bq : bv;
    float*          outp = kqv + (size_t)mat * ND;

    // A fragment: lanes 0-15 rows M=lane, K chunks {0..7,16..23}; hi lanes +8
    const _Float16* arow = xh + (size_t)(n0 + ml) * DIM + hi * 8;
    // B fragment: lane column N=ml -> row o0+ml of W; K chunks {0..7,16..23}; hi lanes +8
    const _Float16* brow = wm + (size_t)(o0 + ml) * DIM + hi * 8;

    v8f c = {};
#pragma unroll
    for (int k0 = 0; k0 < DIM; k0 += 32) {
        v8h a0 = *(const v8h*)(arow + k0);
        v8h a1 = *(const v8h*)(arow + k0 + 16);
        v16h a;
#pragma unroll
        for (int i = 0; i < 8; i++) { a[i] = a0[i]; a[8 + i] = a1[i]; }
        v8h b0 = *(const v8h*)(brow + k0);
        v8h b1 = *(const v8h*)(brow + k0 + 16);
        v16h b;
#pragma unroll
        for (int i = 0; i < 8; i++) { b[i] = b0[i]; b[8 + i] = b1[i]; }
        c = __builtin_amdgcn_wmma_f32_16x16x32_f16(
                /*neg_a=*/false, a, /*neg_b=*/false, b,
                /*c_mod=*/(short)0, c, /*reuse_a=*/false, /*reuse_b=*/false);
    }

    float bias_v = bias[o0 + ml];                 // column fixed per lane
    float* orow  = outp + (size_t)(n0 + hi * 8) * DIM + (o0 + ml);
#pragma unroll
    for (int r = 0; r < 8; r++) orow[(size_t)r * DIM] = c[r] + bias_v;
}

// ------------------------------------------- edge attention logits + seg-max
// One wave per edge; lane l owns dims [4l,4l+4), head h = l>>2.
__global__ void k_edge_att(const float* __restrict__ kbuf,
                           const float* __restrict__ qbuf,
                           const int*   __restrict__ src,
                           const int*   __restrict__ dst,
                           const float* __restrict__ ew,
                           const float* __restrict__ rel_att,  // [8][16]
                           float* __restrict__ att,            // [E][8]
                           unsigned* __restrict__ mbuf)        // [N][8] encoded
{
    int e = blockIdx.x * (blockDim.x >> 5) + (threadIdx.x >> 5);
    if (e >= N_EDGES) return;
    int l  = threadIdx.x & 31;
    int s_ = src[e], d_ = dst[e];
    float w = ew[e];
    float4 qv = *(const float4*)(qbuf + (size_t)d_ * DIM + 4 * l);
    float4 kv = *(const float4*)(kbuf + (size_t)s_ * DIM + 4 * l);
    float4 ra = *(const float4*)(rel_att + 4 * l);   // h*16 + 4*(l&3) == 4*l
    float p = (qv.x + w * ra.x) * kv.x + (qv.y + w * ra.y) * kv.y +
              (qv.z + w * ra.z) * kv.z + (qv.w + w * ra.w) * kv.w;
    p += __shfl_xor(p, 1, 32);
    p += __shfl_xor(p, 2, 32);
    if ((l & 3) == 0) {
        int h = l >> 2;
        float a = p * 0.25f;                         // 1/sqrt(D_K)
        att[(size_t)e * 8 + h] = a;
        atomicMax(mbuf + (size_t)d_ * 8 + h, enc_f32(a));
    }
}

// --------------------------------------------------- exp(att-m) + seg-sum
__global__ void k_edge_exp(const int* __restrict__ dst,
                           const unsigned* __restrict__ mbuf,
                           float* __restrict__ att,
                           float* __restrict__ sbuf)
{
    int i = blockIdx.x * blockDim.x + threadIdx.x;
    if (i >= (int)E8) return;
    int e = i >> 3, h = i & 7;
    int d_ = dst[e];
    float mval = dec_f32(mbuf[(size_t)d_ * 8 + h]);
    float ev = __expf(att[i] - mval);
    att[i] = ev;
    atomicAdd(sbuf + (size_t)d_ * 8 + h, ev);
}

// ------------------------- message matvec (v @ rel_msg) + weighted scatter
__global__ void k_edge_agg(const float* __restrict__ vbuf,
                           const int*   __restrict__ src,
                           const int*   __restrict__ dst,
                           const float* __restrict__ att,    // holds exp weights
                           const float* __restrict__ sbuf,
                           const float* __restrict__ rel_msg, // [8][16][16]
                           float* __restrict__ out)
{
    __shared__ float RM[N_HEADS * D_K * 17];   // stride-17 pad vs bank conflicts
    for (int t = threadIdx.x; t < N_HEADS * D_K * D_K; t += blockDim.x) {
        int h = t >> 8, rem = t & 255, d = rem >> 4, kk = rem & 15;
        RM[h * (D_K * 17) + d * 17 + kk] = rel_msg[t];
    }
    __syncthreads();

    int e = blockIdx.x * (blockDim.x >> 5) + (threadIdx.x >> 5);
    if (e >= N_EDGES) return;
    int l  = threadIdx.x & 31;
    int h  = l >> 2, m4 = l & 3;
    int s_ = src[e], d_ = dst[e];

    float4 v4  = *(const float4*)(vbuf + (size_t)s_ * DIM + 4 * l);
    float  wgt = att[(size_t)e * 8 + h] / sbuf[(size_t)d_ * 8 + h];

    const float* rmh = &RM[h * (D_K * 17) + 4 * m4];  // column base for this lane
    float a0 = 0.f, a1 = 0.f, a2 = 0.f, a3 = 0.f;
#pragma unroll
    for (int x = 0; x < 4; x++) {                     // butterfly over head group
        float4 o4;
        if (x == 0) { o4 = v4; }
        else {
            o4.x = __shfl_xor(v4.x, x, 32);
            o4.y = __shfl_xor(v4.y, x, 32);
            o4.z = __shfl_xor(v4.z, x, 32);
            o4.w = __shfl_xor(v4.w, x, 32);
        }
        int dbase = 4 * (m4 ^ x);                     // which v-dims o4 carries
        const float* rp = rmh + dbase * 17;
        a0 += o4.x * rp[0];        a1 += o4.x * rp[1];
        a2 += o4.x * rp[2];        a3 += o4.x * rp[3];
        a0 += o4.y * rp[17 + 0];   a1 += o4.y * rp[17 + 1];
        a2 += o4.y * rp[17 + 2];   a3 += o4.y * rp[17 + 3];
        a0 += o4.z * rp[34 + 0];   a1 += o4.z * rp[34 + 1];
        a2 += o4.z * rp[34 + 2];   a3 += o4.z * rp[34 + 3];
        a0 += o4.w * rp[51 + 0];   a1 += o4.w * rp[51 + 1];
        a2 += o4.w * rp[51 + 2];   a3 += o4.w * rp[51 + 3];
    }
    float* op = out + (size_t)d_ * DIM + 4 * l;       // h*16 + 4*m4 == 4*l
    atomicAdd(op + 0, wgt * a0);
    atomicAdd(op + 1, wgt * a1);
    atomicAdd(op + 2, wgt * a2);
    atomicAdd(op + 3, wgt * a3);
}

__global__ void k_relu(float* __restrict__ p, int n) {
    int i = blockIdx.x * blockDim.x + threadIdx.x;
    if (i < n) p[i] = fmaxf(p[i], 0.f);
}

// =========================================================================
extern "C" void kernel_launch(void* const* d_in, const int* in_sizes, int n_in,
                              void* d_out, int out_size, void* d_ws, size_t ws_size,
                              hipStream_t stream)
{
    const float* x       = (const float*)d_in[0];
    const int*   src     = (const int*)  d_in[1];
    const int*   dst     = (const int*)  d_in[2];
    const float* ew      = (const float*)d_in[3];
    const float* Wk      = (const float*)d_in[4];
    const float* bk      = (const float*)d_in[5];
    const float* Wq      = (const float*)d_in[6];
    const float* bq      = (const float*)d_in[7];
    const float* Wv      = (const float*)d_in[8];
    const float* bv      = (const float*)d_in[9];
    const float* rel_att = (const float*)d_in[10];
    const float* rel_msg = (const float*)d_in[11];
    float*       out     = (float*)d_out;

    // workspace carving (256 B aligned)
    char*  ws  = (char*)d_ws;
    size_t off = 0;
    auto carve = [&](size_t bytes) -> char* {
        char* p = ws + off;
        off = (off + bytes + 255) & ~(size_t)255;
        return p;
    };
    _Float16* xh   = (_Float16*)carve(ND * sizeof(_Float16));
    _Float16* wh   = (_Float16*)carve(3 * DIM * DIM * sizeof(_Float16));
    float*    kqv  = (float*)   carve(3 * ND * sizeof(float));
    float*    att  = (float*)   carve(E8 * sizeof(float));
    unsigned* mbuf = (unsigned*)carve(N8 * sizeof(unsigned));
    float*    sbuf = (float*)   carve(N8 * sizeof(float));
    (void)ws_size; (void)in_sizes; (void)n_in; (void)out_size;

    const int B = 256;

    // precision conversion (f32 -> f16 for WMMA operands)
    k_f32_to_f16<<<(int)((ND + B - 1) / B), B, 0, stream>>>(x,  xh, (int)ND);
    k_f32_to_f16<<<64, B, 0, stream>>>(Wk, wh,               DIM * DIM);
    k_f32_to_f16<<<64, B, 0, stream>>>(Wq, wh + DIM * DIM,   DIM * DIM);
    k_f32_to_f16<<<64, B, 0, stream>>>(Wv, wh + 2 * DIM * DIM, DIM * DIM);

    // init accumulators every call (graph-replay safe)
    k_zero_u32<<<(int)((ND + B - 1) / B), B, 0, stream>>>((unsigned*)out, (int)ND);
    k_zero_u32<<<(int)((N8 + B - 1) / B), B, 0, stream>>>(mbuf, (int)N8);
    k_zero_u32<<<(int)((N8 + B - 1) / B), B, 0, stream>>>((unsigned*)sbuf, (int)N8);

    // k/q/v projections: 3 * 3125 * 8 = 75000 waves, 8 waves / block
    k_gemm_qkv<<<9375, B, 0, stream>>>(xh, wh, bk, bq, bv, kqv);

    const float* kbuf = kqv;
    const float* qbuf = kqv + ND;
    const float* vbuf = kqv + 2 * ND;

    // edge phase: wave per edge (800000 / 8 waves per block)
    k_edge_att<<<100000, B, 0, stream>>>(kbuf, qbuf, src, dst, ew, rel_att,
                                         att, mbuf);
    k_edge_exp<<<(int)((E8 + B - 1) / B), B, 0, stream>>>(dst, mbuf, att, sbuf);
    k_edge_agg<<<100000, B, 0, stream>>>(vbuf, src, dst, att, sbuf, rel_msg, out);

    k_relu<<<(int)((ND + B - 1) / B), B, 0, stream>>>(out, (int)ND);
}
